// EGAT_30442728194376
// MI455X (gfx1250) — matrix operand order repaired and checked by reading, not compiled, not run on hardware
//
#include <hip/hip_runtime.h>
#include <hip/hip_bf16.h>
#include <math.h>

#define NN 30000
#define NE 480000
// L1: K=192 (3*64), O=128 (4*32);  L2: K=384 (3*128), O=256 (4*64)

typedef __attribute__((ext_vector_type(8)))  float          v8f;
typedef __attribute__((ext_vector_type(8)))  unsigned int   v8u;
typedef __attribute__((ext_vector_type(16))) __bf16         v16bf;

__device__ __forceinline__ unsigned short f2bf(float f) {
    unsigned int x = __float_as_uint(f);
    x += 0x7FFFu + ((x >> 16) & 1u);          // round-to-nearest-even
    return (unsigned short)(x >> 16);
}

// ---------------- elementwise / setup kernels ----------------

__global__ void fill_kernel(float* __restrict__ p, float v, int n) {
    int i = blockIdx.x * blockDim.x + threadIdx.x;
    if (i < n) p[i] = v;
}

// y[i] = bf16(x[i])
__global__ void cvt_bf16(const float* __restrict__ x, unsigned short* __restrict__ y, int n) {
    int i = blockIdx.x * blockDim.x + threadIdx.x;
    if (i < n) y[i] = f2bf(x[i]);
}

// Pack W [O, K] row-major into WMMA B-fragments (bf16):
// frag (t = n-tile, ks = k-step) ; lane j holds col = t*16 + (j%16),
// element i holds k = ks*32 + (i<8 ? 8*(j/16) + i : 16 + 8*(j/16) + (i-8))
__global__ void pack_w(const float* __restrict__ W, unsigned short* __restrict__ Wp,
                       int NT, int KS, int Kin) {
    int id = blockIdx.x * blockDim.x + threadIdx.x;
    int total = NT * KS * 32 * 16;
    if (id >= total) return;
    int i    = id & 15;
    int lane = (id >> 4) & 31;
    int fi   = id >> 9;
    int ks   = fi % KS;
    int t    = fi / KS;
    int col  = t * 16 + (lane & 15);
    int k    = ks * 32 + ((i < 8) ? ((lane >> 4) * 8 + i) : (16 + (lane >> 4) * 8 + (i - 8)));
    Wp[id]   = f2bf(W[(size_t)col * Kin + k]);
}

// wsum[f] = sum_h Wa[h][f]
__global__ void wsum_kernel(const float* __restrict__ Wa, float* __restrict__ ws,
                            int H, int F) {
    int f = blockIdx.x * blockDim.x + threadIdx.x;
    if (f >= F) return;
    float s = 0.f;
    for (int h = 0; h < H; ++h) s += Wa[h * F + f];
    ws[f] = s;
}

// hn1[dst] += efeats[e]  (segment_sum, layer-1 "first_layer" aggregation)
__global__ void edge_agg(const float* __restrict__ ef, const int* __restrict__ dst,
                         float* __restrict__ hn1) {
    int id = blockIdx.x * blockDim.x + threadIdx.x;   // e*16 + q (q = group of 4 feats)
    if (id >= NE * 16) return;
    int e = id >> 4, q = id & 15;
    float4 v = ((const float4*)ef)[(size_t)e * 16 + q];
    float* o = hn1 + (size_t)dst[e] * 64 + q * 4;
    atomicAdd(o + 0, v.x); atomicAdd(o + 1, v.y);
    atomicAdd(o + 2, v.z); atomicAdd(o + 3, v.w);
}

// Y[N,O] = X[N,K] @ W[O,K]^T + b   (small node projections, fp32 VALU)
__global__ void node_gemm(const float* __restrict__ X, const float* __restrict__ W,
                          const float* __restrict__ b, float* __restrict__ Y,
                          int N, int K, int O) {
    int id = blockIdx.x * blockDim.x + threadIdx.x;
    if (id >= N * O) return;
    int n = id / O, o = id % O;
    const float* x = X + (size_t)n * K;
    const float* w = W + (size_t)o * K;
    float s = b[o];
    for (int k = 0; k < K; ++k) s += x[k] * w[k];
    Y[id] = s;
}

// y = elu(x) in f32 and bf16 simultaneously
__global__ void elu_dual(const float* __restrict__ x, float* __restrict__ y,
                         unsigned short* __restrict__ yb, int n) {
    int i = blockIdx.x * blockDim.x + threadIdx.x;
    if (i < n) {
        float v = x[i];
        float e = (v > 0.f) ? v : (expf(v) - 1.f);
        y[i] = e;
        yb[i] = f2bf(e);
    }
}

// ---------------- WMMA fragment helper ----------------
// A sources are bf16 rows in memory; a run of 8 consecutive bf16 = one uint4.
// Run 0 -> fragment elements 0..7 (dwords 0..3), run 1 -> 8..15 (dwords 4..7),
// matching the ISA 16-bit A 16x32 layout (lane<16: K{0..7,16..23}; lane>=16: K{8..15,24..31}).

__device__ __forceinline__ v16bf make_afrag(const unsigned short* __restrict__ p0,
                                            const unsigned short* __restrict__ p1) {
    uint4 q0 = *(const uint4*)p0;
    uint4 q1 = *(const uint4*)p1;
    v8u u;
    u[0] = q0.x; u[1] = q0.y; u[2] = q0.z; u[3] = q0.w;
    u[4] = q1.x; u[5] = q1.y; u[6] = q1.z; u[7] = q1.w;
    return __builtin_bit_cast(v16bf, u);
}

// ---------------- layer-1 edge GEMM (WMMA) ----------------
// One wave per 16-edge tile: f_out = leaky_relu([h_src|e|h_dst] @ We1^T) [16,128]
// then a1[e,h] = f_out . wsum (per head), he1 = bf16(elu(f_out)).
__global__ __launch_bounds__(128) void edge_gemm1(
    const unsigned short* __restrict__ hn1b, const unsigned short* __restrict__ efb,
    const int* __restrict__ src, const int* __restrict__ dst,
    const unsigned short* __restrict__ Wp, const float* __restrict__ wsum,
    float* __restrict__ a_out, unsigned short* __restrict__ he_out) {
    __shared__ float lds[4][16][129];
    const int wave = threadIdx.x >> 5;
    const int lane = threadIdx.x & 31;
    const int e0   = (blockIdx.x * 4 + wave) * 16;
    const int e    = e0 + (lane & 15);
    const int s = src[e], d = dst[e];
    const unsigned short* pS = hn1b + (size_t)s * 64;
    const unsigned short* pE = efb  + (size_t)e * 64;
    const unsigned short* pD = hn1b + (size_t)d * 64;

    v8f zero = {0.f,0.f,0.f,0.f,0.f,0.f,0.f,0.f};
    v8f acc[8];
#pragma unroll
    for (int t = 0; t < 8; ++t) acc[t] = zero;

#pragma unroll
    for (int ks = 0; ks < 6; ++ks) {
        int r0 = ks * 32 + ((lane < 16) ? 0  : 8);
        int r1 = ks * 32 + ((lane < 16) ? 16 : 24);
        const unsigned short* p0 =
            (r0 < 64) ? (pS + r0) : ((r0 < 128) ? (pE + r0 - 64) : (pD + r0 - 128));
        const unsigned short* p1 =
            (r1 < 64) ? (pS + r1) : ((r1 < 128) ? (pE + r1 - 64) : (pD + r1 - 128));
        v16bf A = make_afrag(p0, p1);
#pragma unroll
        for (int t = 0; t < 8; ++t) {
            v16bf B = *(const v16bf*)(Wp + ((size_t)(t * 6 + ks) * 32 + lane) * 16);
            acc[t] = __builtin_amdgcn_wmma_f32_16x16x32_bf16(
                false, A, false, B, (short)0, acc[t], false, false);
        }
    }

    const int mhi = (lane < 16) ? 0 : 8;
    const int nl  = lane & 15;
#pragma unroll
    for (int t = 0; t < 8; ++t)
#pragma unroll
        for (int r = 0; r < 8; ++r) {
            float v = acc[t][r];
            lds[wave][r + mhi][t * 16 + nl] = (v > 0.f) ? v : 0.01f * v;   // leaky_relu
        }
    __syncthreads();

    if (lane < 16) {
        const float* rowp = &lds[wave][lane][0];
        float as0 = 0.f, as1 = 0.f, as2 = 0.f, as3 = 0.f;
        unsigned short* ho = he_out + (size_t)(e0 + lane) * 128;
        for (int c = 0; c < 128; ++c) {
            float v  = rowp[c];
            float wv = wsum[c & 31] * v;
            int h = c >> 5;
            if      (h == 0) as0 += wv;
            else if (h == 1) as1 += wv;
            else if (h == 2) as2 += wv;
            else             as3 += wv;
            float ev = (v > 0.f) ? v : (expf(v) - 1.f);                    // elu
            ho[c] = f2bf(ev);
        }
        float* ao = a_out + (size_t)(e0 + lane) * 4;
        ao[0] = as0; ao[1] = as1; ao[2] = as2; ao[3] = as3;
    }
}

// ---------------- layer-2 edge GEMM (WMMA) ----------------
// f_out2 = leaky_relu([hn[src]|he1|hn[dst]] @ We2^T) [16,256];
// a2[e,h] = f_out2 . wsum2 ; he_final[e,:] = mean over 4 heads -> d_out.
__global__ __launch_bounds__(64) void edge_gemm2(
    const unsigned short* __restrict__ hnl1b, const unsigned short* __restrict__ he1,
    const int* __restrict__ src, const int* __restrict__ dst,
    const unsigned short* __restrict__ Wp, const float* __restrict__ wsum,
    float* __restrict__ a_out, float* __restrict__ he_final) {
    __shared__ float lds[2][16][257];
    const int wave = threadIdx.x >> 5;
    const int lane = threadIdx.x & 31;
    const int e0   = (blockIdx.x * 2 + wave) * 16;
    const int e    = e0 + (lane & 15);
    const int s = src[e], d = dst[e];
    const unsigned short* pS = hnl1b + (size_t)s * 128;
    const unsigned short* pE = he1   + (size_t)e * 128;
    const unsigned short* pD = hnl1b + (size_t)d * 128;

    v8f zero = {0.f,0.f,0.f,0.f,0.f,0.f,0.f,0.f};
    v8f acc[16];
#pragma unroll
    for (int t = 0; t < 16; ++t) acc[t] = zero;

    for (int ks = 0; ks < 12; ++ks) {
        int r0 = ks * 32 + ((lane < 16) ? 0  : 8);
        int r1 = ks * 32 + ((lane < 16) ? 16 : 24);
        const unsigned short* p0 =
            (r0 < 128) ? (pS + r0) : ((r0 < 256) ? (pE + r0 - 128) : (pD + r0 - 256));
        const unsigned short* p1 =
            (r1 < 128) ? (pS + r1) : ((r1 < 256) ? (pE + r1 - 128) : (pD + r1 - 256));
        v16bf A = make_afrag(p0, p1);
#pragma unroll
        for (int t = 0; t < 16; ++t) {
            v16bf B = *(const v16bf*)(Wp + ((size_t)(t * 12 + ks) * 32 + lane) * 16);
            acc[t] = __builtin_amdgcn_wmma_f32_16x16x32_bf16(
                false, A, false, B, (short)0, acc[t], false, false);
        }
    }

    const int mhi = (lane < 16) ? 0 : 8;
    const int nl  = lane & 15;
#pragma unroll
    for (int t = 0; t < 16; ++t)
#pragma unroll
        for (int r = 0; r < 8; ++r) {
            float v = acc[t][r];
            lds[wave][r + mhi][t * 16 + nl] = (v > 0.f) ? v : 0.01f * v;   // leaky_relu
        }
    __syncthreads();

    if (lane < 16) {
        const float* rowp = &lds[wave][lane][0];
        float as0 = 0.f, as1 = 0.f, as2 = 0.f, as3 = 0.f;
        float* ho = he_final + (size_t)(e0 + lane) * 64;
        for (int j = 0; j < 64; ++j) {
            float w  = wsum[j];
            float v0 = rowp[j], v1 = rowp[64 + j], v2 = rowp[128 + j], v3 = rowp[192 + j];
            as0 += v0 * w; as1 += v1 * w; as2 += v2 * w; as3 += v3 * w;
            ho[j] = 0.25f * (v0 + v1 + v2 + v3);                            // head mean
        }
        float* ao = a_out + (size_t)(e0 + lane) * 4;
        ao[0] = as0; ao[1] = as1; ao[2] = as2; ao[3] = as3;
    }
}

// ---------------- segment softmax + scatter ----------------

__device__ __forceinline__ void atomicMaxF(float* addr, float v) {
    if (v >= 0.f) atomicMax((int*)addr, __float_as_int(v));
    else          atomicMin((unsigned int*)addr, __float_as_uint(v));
}

__global__ void seg_max(const float* __restrict__ a, const int* __restrict__ dst,
                        float* __restrict__ amax) {
    int id = blockIdx.x * blockDim.x + threadIdx.x;
    if (id >= NE * 4) return;
    int e = id >> 2, h = id & 3;
    atomicMaxF(&amax[(size_t)dst[e] * 4 + h], a[id]);
}

__global__ void seg_den(const float* __restrict__ a, const int* __restrict__ dst,
                        const float* __restrict__ amax, float* __restrict__ den) {
    int id = blockIdx.x * blockDim.x + threadIdx.x;
    if (id >= NE * 4) return;
    int e = id >> 2, h = id & 3;
    int d = dst[e];
    atomicAdd(&den[(size_t)d * 4 + h], expf(a[id] - amax[(size_t)d * 4 + h]));
}

// hout[dst, (h)*W : ...] += alpha * hp[src,h,:W] * scale
__global__ void scatter_msg(const float* __restrict__ a, const int* __restrict__ src,
                            const int* __restrict__ dst, const float* __restrict__ amax,
                            const float* __restrict__ den, const float* __restrict__ hp,
                            float* __restrict__ hout, int W, int dstW, float scale) {
    int id = blockIdx.x * blockDim.x + threadIdx.x;
    if (id >= NE * 4) return;
    int e = id >> 2, h = id & 3;
    int d = dst[e], s = src[e];
    float dn = den[(size_t)d * 4 + h];
    dn = (dn > 1e-9f) ? dn : 1e-9f;
    float alpha = scale * expf(a[id] - amax[(size_t)d * 4 + h]) / dn;
    const float* hs = hp + (size_t)s * (4 * W) + h * W;
    float* o = hout + (size_t)d * dstW + ((dstW == 4 * W) ? h * W : 0);
    for (int j = 0; j < W; ++j) atomicAdd(o + j, alpha * hs[j]);
}

// ---------------- launcher ----------------

extern "C" void kernel_launch(void* const* d_in, const int* in_sizes, int n_in,
                              void* d_out, int out_size, void* d_ws, size_t ws_size,
                              hipStream_t stream) {
    (void)in_sizes; (void)n_in; (void)out_size; (void)ws_size;
    const float* efeats = (const float*)d_in[1];   // nfeats (d_in[0]) is dead in reference
    const int*   src    = (const int*)d_in[2];
    const int*   dst    = (const int*)d_in[3];
    const float* Wn1 = (const float*)d_in[4];
    const float* b1  = (const float*)d_in[5];
    const float* We1 = (const float*)d_in[6];
    const float* Wa1 = (const float*)d_in[7];
    const float* Wn2 = (const float*)d_in[8];
    const float* b2  = (const float*)d_in[9];
    const float* We2 = (const float*)d_in[10];
    const float* Wa2 = (const float*)d_in[11];

    float* out_nodes = (float*)d_out;                       // [NN,64]
    float* out_edges = out_nodes + (size_t)NN * 64;         // [NE,64]

    char* w = (char*)d_ws;
    auto take = [&](size_t bytes) -> char* {
        char* p = w; w += (bytes + 255) & ~(size_t)255; return p;
    };
    float* hn1   = (float*)take((size_t)NN * 64 * 4);
    float* hp1   = (float*)take((size_t)NN * 128 * 4);
    float* hout1 = (float*)take((size_t)NN * 128 * 4);
    float* hnl1  = (float*)take((size_t)NN * 128 * 4);
    float* hp2   = (float*)take((size_t)NN * 256 * 4);
    float* a1    = (float*)take((size_t)NE * 4 * 4);
    float* a2    = (float*)take((size_t)NE * 4 * 4);
    float* amax1 = (float*)take((size_t)NN * 4 * 4);
    float* den1  = (float*)take((size_t)NN * 4 * 4);
    float* amax2 = (float*)take((size_t)NN * 4 * 4);
    float* den2  = (float*)take((size_t)NN * 4 * 4);
    unsigned short* he1   = (unsigned short*)take((size_t)NE * 128 * 2);
    unsigned short* efb   = (unsigned short*)take((size_t)NE * 64 * 2);
    unsigned short* hn1b  = (unsigned short*)take((size_t)NN * 64 * 2);
    unsigned short* hnl1b = (unsigned short*)take((size_t)NN * 128 * 2);
    unsigned short* w1p   = (unsigned short*)take((size_t)8 * 6 * 32 * 16 * 2);
    unsigned short* w2p   = (unsigned short*)take((size_t)16 * 12 * 32 * 16 * 2);
    float* ws1 = (float*)take(32 * 4);
    float* ws2 = (float*)take(64 * 4);

    const int B = 256;
    auto blk = [&](long n) { return (int)((n + B - 1) / B); };

    // init accumulators
    fill_kernel<<<blk((long)NN * 64),  B, 0, stream>>>(hn1,   0.f, NN * 64);
    fill_kernel<<<blk((long)NN * 128), B, 0, stream>>>(hout1, 0.f, NN * 128);
    fill_kernel<<<blk((long)NN * 4),   B, 0, stream>>>(den1,  0.f, NN * 4);
    fill_kernel<<<blk((long)NN * 4),   B, 0, stream>>>(den2,  0.f, NN * 4);
    fill_kernel<<<blk((long)NN * 4),   B, 0, stream>>>(amax1, -INFINITY, NN * 4);
    fill_kernel<<<blk((long)NN * 4),   B, 0, stream>>>(amax2, -INFINITY, NN * 4);
    fill_kernel<<<blk((long)NN * 64),  B, 0, stream>>>(out_nodes, 0.f, NN * 64);

    // weight / operand prep
    pack_w<<<blk(8L * 6 * 32 * 16),   B, 0, stream>>>(We1, w1p, 8, 6, 192);
    pack_w<<<blk(16L * 12 * 32 * 16), B, 0, stream>>>(We2, w2p, 16, 12, 384);
    wsum_kernel<<<1, 32, 0, stream>>>(Wa1, ws1, 4, 32);
    wsum_kernel<<<1, 64, 0, stream>>>(Wa2, ws2, 4, 64);
    cvt_bf16<<<blk((long)NE * 64), B, 0, stream>>>(efeats, efb, NE * 64);

    // ---- layer 1 ----
    edge_agg<<<blk((long)NE * 16), B, 0, stream>>>(efeats, dst, hn1);
    cvt_bf16<<<blk((long)NN * 64), B, 0, stream>>>(hn1, hn1b, NN * 64);
    node_gemm<<<blk((long)NN * 128), B, 0, stream>>>(hn1, Wn1, b1, hp1, NN, 64, 128);
    edge_gemm1<<<NE / 16 / 4, 128, 0, stream>>>(hn1b, efb, src, dst, w1p, ws1, a1, he1);
    seg_max<<<blk((long)NE * 4), B, 0, stream>>>(a1, dst, amax1);
    seg_den<<<blk((long)NE * 4), B, 0, stream>>>(a1, dst, amax1, den1);
    scatter_msg<<<blk((long)NE * 4), B, 0, stream>>>(a1, src, dst, amax1, den1, hp1,
                                                     hout1, 32, 128, 1.f);
    elu_dual<<<blk((long)NN * 128), B, 0, stream>>>(hout1, hnl1, hnl1b, NN * 128);

    // ---- layer 2 ----
    node_gemm<<<blk((long)NN * 256), B, 0, stream>>>(hnl1, Wn2, b2, hp2, NN, 128, 256);
    edge_gemm2<<<NE / 16 / 2, 64, 0, stream>>>(hnl1b, he1, src, dst, w2p, ws2, a2, out_edges);
    seg_max<<<blk((long)NE * 4), B, 0, stream>>>(a2, dst, amax2);
    seg_den<<<blk((long)NE * 4), B, 0, stream>>>(a2, dst, amax2, den2);
    // head-mean folded into scale = 0.25, accumulated directly into d_out node block
    scatter_msg<<<blk((long)NE * 4), B, 0, stream>>>(a2, src, dst, amax2, den2, hp2,
                                                     out_nodes, 64, 64, 0.25f);
}
